// MultiHeadLatentAttention_45861660786898
// MI455X (gfx1250) — matrix operand order repaired and checked by reading, not compiled, not run on hardware
//
#include <hip/hip_runtime.h>
#include <cmath>

// ---------------------------------------------------------------------------
// MLA forward for MI455X (gfx1250, wave32, WMMA f32_16x16x32_bf16 + TDM)
// ---------------------------------------------------------------------------

typedef __attribute__((ext_vector_type(16))) __bf16 v16bf;
typedef __attribute__((ext_vector_type(8)))  float  v8f;
typedef __attribute__((ext_vector_type(4)))  unsigned int v4u;
typedef __attribute__((ext_vector_type(8)))  int v8i;
typedef __attribute__((ext_vector_type(4)))  int v4i;

#define L_SEQ     4096
#define DMODEL    2048
#define NHEADS    16
#define HDIM      128
#define CDIM      512

#if defined(__has_builtin)
#if __has_builtin(__builtin_amdgcn_tensor_load_to_lds) && __has_builtin(__builtin_amdgcn_s_wait_tensorcnt)
#define HAVE_TDM 1
#endif
#endif
#ifndef HAVE_TDM
#define HAVE_TDM 0
#endif

union Frag {
    v16bf v;
    unsigned int u[8];
};

__device__ __forceinline__ unsigned short f2bf(float f) {
    unsigned int u = __float_as_uint(f);
    u += 0x7FFFu + ((u >> 16) & 1u);           // round-to-nearest-even
    return (unsigned short)(u >> 16);
}
__device__ __forceinline__ float bf2f(unsigned short h) {
    return __uint_as_float(((unsigned int)h) << 16);
}

#if HAVE_TDM
// TDM: DMA a 128-row x 32-elem bf16 tile (row stride K elems) from global to
// LDS, adding 16B of pad after every 64B row -> LDS row stride 40 elems.
// D# per CDNA5 ISA; 6-arg builtin form
// (clang-23): (g0 x4, g1 x8, g2 x4, g3 x4, extra x8, cpol).
__device__ __forceinline__ void tdm_load_tile_a(const unsigned short* gptr,
                                                unsigned lds_off, int K) {
    unsigned long long ga = (unsigned long long)(__UINTPTR_TYPE__)gptr;
    v4u g0;
    g0.x = 1u;                                              // count=1 (valid), 2D, no gather
    g0.y = lds_off;                                         // lds_addr [63:32]
    g0.z = (unsigned)(ga & 0xFFFFFFFFu);                    // global_addr [95:64]
    g0.w = 0x80000000u | (unsigned)((ga >> 32) & 0x01FFFFFFu); // addr[56:32] | type=2
    v8i g1;
    // data_size=1 (2B) | pad_enable | pad_interval=3 (16 dw = 64B) | pad_amount=3 (4 dw = 16B)
    g1[0] = (int)((1u << 16) | (1u << 20) | (3u << 22) | (3u << 25));
    g1[1] = (int)(((unsigned)K & 0xFFFFu) << 16);           // tensor_dim0 lo16 @ [63:48]
    g1[2] = (int)((((unsigned)K >> 16) & 0xFFFFu) | (128u << 16)); // dim0 hi | tensor_dim1 lo16
    g1[3] = (int)(32u << 16);                               // tensor_dim1 hi=0 | tile_dim0=32
    g1[4] = (int)(128u);                                    // tile_dim1=128 | tile_dim2=0
    g1[5] = (int)(unsigned)K;                               // tensor_dim0_stride lo32
    g1[6] = 0;                                              // stride0 hi | stride1 lo
    g1[7] = 0;                                              // stride1 hi
    v4i gz4 = {0, 0, 0, 0};
    v8i gz8 = {0, 0, 0, 0, 0, 0, 0, 0};
    __builtin_amdgcn_tensor_load_to_lds(g0, g1, gz4, gz4, gz8, 0);
}
#endif

// ---------------------------------------------------------------------------
// fp32 -> bf16 cast
// ---------------------------------------------------------------------------
__global__ void cast_f32_bf16_kernel(const float* __restrict__ in,
                                     unsigned short* __restrict__ out, int n) {
    int i = blockIdx.x * blockDim.x + threadIdx.x;
    if (i < n) out[i] = f2bf(in[i]);
}

// ---------------------------------------------------------------------------
// GEMM: C[M,N] = A[M,K](bf16, row-major) * B[K,N](bf16, row-major) (+bias)
// 128x128 block tile, 8 waves (4 m-rows x 2 n-cols), each wave 32x64,
// K-step 32. A tile DMA'd by the Tensor Data Mover (double-buffered, wave 0
// issues, TENSORcnt-gated); B staged transposed [n][k] so fragment loads are
// contiguous dword ds_loads.
// ---------------------------------------------------------------------------
template <bool OUT_BF16, bool HAS_BIAS>
__global__ __launch_bounds__(256)
void gemm_bf16_kernel(const unsigned short* __restrict__ A,
                      const unsigned short* __restrict__ B,
                      void* __restrict__ Cout,
                      const float* __restrict__ bias,
                      int M, int N, int K) {
    __shared__ unsigned short As[2][128 * 40];  // stride 40 (pad 8) -> 80B rows
    __shared__ unsigned short Bt[128 * 40];

    const int tid  = threadIdx.x;
    const int lane = tid & 31;
    const int w    = tid >> 5;
    const int wrow = w & 3;          // 4 wave rows * 32
    const int wcol = w >> 2;         // 2 wave cols * 64
    const int half = lane >> 4;
    const int lm   = lane & 15;
    const int m0   = blockIdx.y * 128;
    const int n0   = blockIdx.x * 128;

    v8f acc[2][4];
#pragma unroll
    for (int mi = 0; mi < 2; ++mi)
#pragma unroll
        for (int ni = 0; ni < 4; ++ni) acc[mi][ni] = {};

#if HAVE_TDM
    const unsigned ldsA0 = (unsigned)(__UINTPTR_TYPE__)&As[0][0];
    const unsigned ldsA1 = (unsigned)(__UINTPTR_TYPE__)&As[1][0];
    if (w == 0)  // prologue: DMA first A tile into buffer 0
        tdm_load_tile_a(A + (size_t)m0 * K, ldsA0, K);
#endif

    int buf = 0;
    for (int kk = 0; kk < K; kk += 32, buf ^= 1) {
#if HAVE_TDM
        if (w == 0) {
            if (kk + 32 < K) {  // DMA next tile into other buffer, then wait
                tdm_load_tile_a(A + (size_t)m0 * K + kk + 32,
                                buf ? ldsA0 : ldsA1, K);
                __builtin_amdgcn_s_wait_tensorcnt(1);  // current tile landed
            } else {
                __builtin_amdgcn_s_wait_tensorcnt(0);
            }
        }
#else
        // --- manual stage A tile 128x32 (uint4 per thread, 2 passes) ---
#pragma unroll
        for (int it = 0; it < 2; ++it) {
            int idx = tid + it * 256;
            int row = idx >> 2;
            int cg  = idx & 3;
            uint4 d = *(const uint4*)(A + (size_t)(m0 + row) * K + kk + cg * 8);
            *(uint4*)(&As[buf][row * 40 + cg * 8]) = d;
        }
#endif
        // --- stage B tile 32x128 transposed into Bt[n][k] ---
#pragma unroll
        for (int it = 0; it < 2; ++it) {
            int idx = tid + it * 256;
            int kr  = idx >> 4;
            int ng  = idx & 15;
            uint4 d = *(const uint4*)(B + (size_t)(kk + kr) * N + n0 + ng * 8);
            unsigned int dv[4] = {d.x, d.y, d.z, d.w};
#pragma unroll
            for (int u = 0; u < 4; ++u) {
                Bt[(ng * 8 + 2 * u    ) * 40 + kr] = (unsigned short)(dv[u] & 0xFFFFu);
                Bt[(ng * 8 + 2 * u + 1) * 40 + kr] = (unsigned short)(dv[u] >> 16);
            }
        }
        __syncthreads();

        if (kk + 32 < K)  // prefetch next B tile -> global_prefetch_b8
            __builtin_prefetch(B + (size_t)(kk + 32 + (tid >> 4)) * N + n0, 0, 1);

        // --- fragments from LDS ---
        Frag a[2], b[4];
#pragma unroll
        for (int mi = 0; mi < 2; ++mi) {
            int row = wrow * 32 + mi * 16 + lm;
#pragma unroll
            for (int j = 0; j < 8; ++j) {
                int k = ((j >> 2) << 4) + ((j & 3) << 1) + (half << 3);
                a[mi].u[j] = *(const unsigned int*)(&As[buf][row * 40 + k]);
            }
        }
#pragma unroll
        for (int ni = 0; ni < 4; ++ni) {
            int n = wcol * 64 + ni * 16 + lm;
#pragma unroll
            for (int j = 0; j < 8; ++j) {
                int k = ((j >> 2) << 4) + ((j & 3) << 1) + (half << 3);
                b[ni].u[j] = *(const unsigned int*)(&Bt[n * 40 + k]);
            }
        }
#pragma unroll
        for (int mi = 0; mi < 2; ++mi)
#pragma unroll
            for (int ni = 0; ni < 4; ++ni)
                acc[mi][ni] = __builtin_amdgcn_wmma_f32_16x16x32_bf16(
                    false, a[mi].v, false, b[ni].v, (short)0, acc[mi][ni], false, false);
        __syncthreads();
    }

    // --- epilogue: D layout m = r + 8*half, n = lm ---
#pragma unroll
    for (int mi = 0; mi < 2; ++mi) {
#pragma unroll
        for (int ni = 0; ni < 4; ++ni) {
#pragma unroll
            for (int r = 0; r < 8; ++r) {
                int m = m0 + wrow * 32 + mi * 16 + r + (half << 3);
                int n = n0 + wcol * 64 + ni * 16 + lm;
                float v = acc[mi][ni][r];
                if (HAS_BIAS) v += bias[n];
                if (OUT_BF16)
                    ((unsigned short*)Cout)[(size_t)m * N + n] = f2bf(v);
                else
                    ((float*)Cout)[(size_t)m * N + n] = v;
            }
        }
    }
}

// ---------------------------------------------------------------------------
// RoPE + pack: base[L,H*64] (bf16) and rot[L,H*64] (bf16) -> out[L,H,128] bf16
// rot part: x1=rot[0:32], x2=rot[32:64]; out=[x1*c - x2*s, x2*c + x1*s]
// ---------------------------------------------------------------------------
__global__ void rope_pack_kernel(const unsigned short* __restrict__ base_in,
                                 const unsigned short* __restrict__ rot_in,
                                 unsigned short* __restrict__ out,
                                 float scale, int total) {
    int t = blockIdx.x * blockDim.x + threadIdx.x;
    if (t >= total) return;
    int i   = t & 31;
    int h   = (t >> 5) & (NHEADS - 1);
    int row = t >> 9;

    const unsigned short* bp = base_in + (size_t)row * (NHEADS * 64) + h * 64;
    const unsigned short* rp = rot_in  + (size_t)row * (NHEADS * 64) + h * 64;
    unsigned short* op = out + ((size_t)row * NHEADS + h) * HDIM;

    op[2 * i]     = f2bf(bf2f(bp[2 * i])     * scale);
    op[2 * i + 1] = f2bf(bf2f(bp[2 * i + 1]) * scale);

    float x1 = bf2f(rp[i]);
    float x2 = bf2f(rp[32 + i]);
    float invf = __expf(-(float)i * 0.28782313662425575f);  // ln(10000)/32
    float ang  = (float)row * invf;
    float c = __cosf(ang);
    float s = __sinf(ang);
    op[64 + i] = f2bf((x1 * c - x2 * s) * scale);
    op[96 + i] = f2bf((x2 * c + x1 * s) * scale);
}

// ---------------------------------------------------------------------------
// V transpose: v[L, H*128] bf16 -> Vt[H*128, L] bf16 (d-major, key contiguous)
// ---------------------------------------------------------------------------
__global__ void transpose_v_kernel(const unsigned short* __restrict__ v,
                                   unsigned short* __restrict__ vt, int total) {
    int t = blockIdx.x * blockDim.x + threadIdx.x;
    if (t >= total) return;
    int row = t >> 11;        // / 2048
    int col = t & 2047;       // h*128 + d
    vt[(size_t)col * L_SEQ + row] = v[t];
}

// ---------------------------------------------------------------------------
// Flash attention: one wave per (16-query tile, head).
// S = Q*K^T via WMMA, online softmax with 16-lane shuffle reductions,
// P staged through LDS (D-layout -> A-layout), PV via WMMA from Vt[H,128,L].
// ---------------------------------------------------------------------------
__global__ __launch_bounds__(32)
void mla_attention_kernel(const unsigned short* __restrict__ Q,   // [L,H,128]
                          const unsigned short* __restrict__ Km,  // [L,H,128]
                          const unsigned short* __restrict__ Vt,  // [H,128,L]
                          unsigned short* __restrict__ ctx) {     // [L, H*128]
    __shared__ unsigned short Plds[16 * 32];

    const int lane  = threadIdx.x;
    const int half  = lane >> 4;
    const int lm    = lane & 15;
    const int qrow0 = blockIdx.x * 16;
    const int head  = blockIdx.y;

    // Q A-fragments for 4 k-chunks of 32 dims (Q already scaled by 1/sqrt(128))
    Frag aq[4];
#pragma unroll
    for (int c = 0; c < 4; ++c) {
        const unsigned short* qp =
            Q + ((size_t)(qrow0 + lm) * NHEADS + head) * HDIM + c * 32;
#pragma unroll
        for (int j = 0; j < 8; ++j) {
            int k = ((j >> 2) << 4) + ((j & 3) << 1) + (half << 3);
            aq[c].u[j] = *(const unsigned int*)(qp + k);
        }
    }

    v8f acc[8];
#pragma unroll
    for (int c = 0; c < 8; ++c) acc[c] = {};
    float mrun[8], lrun[8];
#pragma unroll
    for (int r = 0; r < 8; ++r) { mrun[r] = -1e30f; lrun[r] = 0.0f; }

    const int nchunks = (qrow0 + 16 + 31) >> 5;
    for (int kc = 0; kc < nchunks; ++kc) {
        const int kbase = kc * 32;
        float sv[2][8];
        // ---- scores for two 16-key groups ----
#pragma unroll
        for (int kg = 0; kg < 2; ++kg) {
            int key0 = kbase + kg * 16;
            v8f s = {};
#pragma unroll
            for (int c = 0; c < 4; ++c) {
                Frag bk;
                const unsigned short* kp =
                    Km + ((size_t)(key0 + lm) * NHEADS + head) * HDIM + c * 32;
#pragma unroll
                for (int j = 0; j < 8; ++j) {
                    int k = ((j >> 2) << 4) + ((j & 3) << 1) + (half << 3);
                    bk.u[j] = *(const unsigned int*)(kp + k);
                }
                s = __builtin_amdgcn_wmma_f32_16x16x32_bf16(
                    false, aq[c].v, false, bk.v, (short)0, s, false, false);
            }
#pragma unroll
            for (int r = 0; r < 8; ++r) {
                int q   = qrow0 + r + (half << 3);
                int key = key0 + lm;
                float x = s[r];
                if (key > q) x = -1e30f;       // causal mask
                sv[kg][r] = x;
            }
        }
        // ---- online softmax update ----
        float mnew[8], scalef[8];
#pragma unroll
        for (int r = 0; r < 8; ++r) {
            float mx = fmaxf(sv[0][r], sv[1][r]);
#pragma unroll
            for (int off = 1; off < 16; off <<= 1)
                mx = fmaxf(mx, __shfl_xor(mx, off, 32));
            mnew[r]   = fmaxf(mrun[r], mx);
            scalef[r] = __expf(mrun[r] - mnew[r]);
            mrun[r]   = mnew[r];
        }
#pragma unroll
        for (int kg = 0; kg < 2; ++kg) {
#pragma unroll
            for (int r = 0; r < 8; ++r) {
                float p = __expf(sv[kg][r] - mnew[r]);
                sv[kg][r] = p;
                Plds[(r + (half << 3)) * 32 + kg * 16 + lm] = f2bf(p);
            }
        }
#pragma unroll
        for (int r = 0; r < 8; ++r) {
            float psum = sv[0][r] + sv[1][r];
#pragma unroll
            for (int off = 1; off < 16; off <<= 1)
                psum += __shfl_xor(psum, off, 32);
            lrun[r] = lrun[r] * scalef[r] + psum;
        }
        __syncthreads();
        // ---- P as A-fragment ----
        Frag ap;
#pragma unroll
        for (int j = 0; j < 8; ++j) {
            int k = ((j >> 2) << 4) + ((j & 3) << 1) + (half << 3);
            ap.u[j] = *(const unsigned int*)(&Plds[lm * 32 + k]);
        }
        // ---- rescale + PV over 8 d-chunks of 16 ----
#pragma unroll
        for (int c = 0; c < 8; ++c) {
            v8f a = acc[c];
#pragma unroll
            for (int r = 0; r < 8; ++r) a[r] *= scalef[r];
            Frag bv;
            const unsigned short* vp =
                Vt + ((size_t)head * HDIM + c * 16 + lm) * L_SEQ + kbase;
#pragma unroll
            for (int j = 0; j < 8; ++j) {
                int k = ((j >> 2) << 4) + ((j & 3) << 1) + (half << 3);
                bv.u[j] = *(const unsigned int*)(vp + k);
            }
            acc[c] = __builtin_amdgcn_wmma_f32_16x16x32_bf16(
                false, ap.v, false, bv.v, (short)0, a, false, false);
        }
        __syncthreads();
    }

    // ---- normalize + write ctx bf16 [L, H*128] ----
#pragma unroll
    for (int r = 0; r < 8; ++r) lrun[r] = 1.0f / lrun[r];
#pragma unroll
    for (int c = 0; c < 8; ++c) {
#pragma unroll
        for (int r = 0; r < 8; ++r) {
            int m = qrow0 + r + (half << 3);
            ctx[(size_t)m * DMODEL + head * HDIM + c * 16 + lm] =
                f2bf(acc[c][r] * lrun[r]);
        }
    }
}

// ---------------------------------------------------------------------------
// Host orchestration
// ---------------------------------------------------------------------------
extern "C" void kernel_launch(void* const* d_in, const int* in_sizes, int n_in,
                              void* d_out, int out_size, void* d_ws, size_t ws_size,
                              hipStream_t stream) {
    (void)in_sizes; (void)n_in; (void)out_size; (void)ws_size;

    const float* x    = (const float*)d_in[0];
    const float* Wdkv = (const float*)d_in[1];
    const float* Wdq  = (const float*)d_in[2];
    const float* Wuk  = (const float*)d_in[3];
    const float* Wuv  = (const float*)d_in[4];
    const float* Wuq  = (const float*)d_in[5];
    const float* Wqr  = (const float*)d_in[6];
    const float* Wkr  = (const float*)d_in[7];
    const float* Wout = (const float*)d_in[8];
    const float* bout = (const float*)d_in[9];

    char* ws = (char*)d_ws;
    const size_t MB = 1024 * 1024;
    unsigned short* xb    = (unsigned short*)(ws + 0);         // 16MB (reused as Vt)
    unsigned short* ckv   = (unsigned short*)(ws + 16 * MB);   // 4MB
    unsigned short* cq    = (unsigned short*)(ws + 20 * MB);   // 4MB
    unsigned short* qbase = (unsigned short*)(ws + 24 * MB);   // 8MB (reused as ctx)
    unsigned short* qr    = (unsigned short*)(ws + 32 * MB);   // 8MB
    unsigned short* kbase = (unsigned short*)(ws + 40 * MB);   // 8MB
    unsigned short* kr    = (unsigned short*)(ws + 48 * MB);   // 8MB
    unsigned short* vbuf  = (unsigned short*)(ws + 56 * MB);   // 16MB
    unsigned short* Qb    = (unsigned short*)(ws + 72 * MB);   // 16MB
    unsigned short* Kb    = (unsigned short*)(ws + 88 * MB);   // 16MB
    unsigned short* wdkvb = (unsigned short*)(ws + 104 * MB);  // 2MB
    unsigned short* wdqb  = (unsigned short*)(ws + 106 * MB);  // 2MB
    unsigned short* wukb  = (unsigned short*)(ws + 108 * MB);  // 1MB
    unsigned short* wuvb  = (unsigned short*)(ws + 109 * MB);  // 2MB
    unsigned short* wuqb  = (unsigned short*)(ws + 111 * MB);  // 1MB
    unsigned short* wqrb  = (unsigned short*)(ws + 112 * MB);  // 1MB
    unsigned short* wkrb  = (unsigned short*)(ws + 113 * MB);  // 1MB
    unsigned short* woutb = (unsigned short*)(ws + 114 * MB);  // 8MB (ends 122MB)
    unsigned short* Vt  = xb;     // xb dead after c_q GEMM (stream order)
    unsigned short* ctx = qbase;  // qbase+qr (16MB) dead after rope_q

    auto cast = [&](const float* src, unsigned short* dst, int n) {
        cast_f32_bf16_kernel<<<(n + 255) / 256, 256, 0, stream>>>(src, dst, n);
    };
    cast(x,    xb,    L_SEQ * DMODEL);
    cast(Wdkv, wdkvb, DMODEL * CDIM);
    cast(Wdq,  wdqb,  DMODEL * CDIM);
    cast(Wuk,  wukb,  CDIM * (NHEADS * 64));
    cast(Wuv,  wuvb,  CDIM * (NHEADS * 128));
    cast(Wuq,  wuqb,  CDIM * (NHEADS * 64));
    cast(Wqr,  wqrb,  CDIM * (NHEADS * 64));
    cast(Wkr,  wkrb,  CDIM * (NHEADS * 64));
    cast(Wout, woutb, DMODEL * DMODEL);

    auto gemm_bf = [&](const unsigned short* A, const unsigned short* B,
                       unsigned short* C, int M, int N, int K) {
        dim3 grid(N / 128, M / 128);
        gemm_bf16_kernel<true, false><<<grid, 256, 0, stream>>>(
            A, B, (void*)C, nullptr, M, N, K);
    };
    gemm_bf(xb,  wdkvb, ckv,   L_SEQ, CDIM, DMODEL);       // c_kv
    gemm_bf(xb,  wdqb,  cq,    L_SEQ, CDIM, DMODEL);       // c_q
    gemm_bf(ckv, wukb,  kbase, L_SEQ, NHEADS * 64,  CDIM); // k_base
    gemm_bf(ckv, wkrb,  kr,    L_SEQ, NHEADS * 64,  CDIM); // k_r
    gemm_bf(ckv, wuvb,  vbuf,  L_SEQ, NHEADS * 128, CDIM); // v
    gemm_bf(cq,  wuqb,  qbase, L_SEQ, NHEADS * 64,  CDIM); // q_base
    gemm_bf(cq,  wqrb,  qr,    L_SEQ, NHEADS * 64,  CDIM); // q_r

    const float qscale = 1.0f / sqrtf((float)HDIM);
    const int ropeTot = L_SEQ * NHEADS * 32;
    rope_pack_kernel<<<(ropeTot + 255) / 256, 256, 0, stream>>>(qbase, qr, Qb, qscale, ropeTot);
    rope_pack_kernel<<<(ropeTot + 255) / 256, 256, 0, stream>>>(kbase, kr, Kb, 1.0f, ropeTot);

    const int vtTot = L_SEQ * DMODEL;
    transpose_v_kernel<<<(vtTot + 255) / 256, 256, 0, stream>>>(vbuf, Vt, vtTot);

    dim3 agrid(L_SEQ / 16, NHEADS);
    mla_attention_kernel<<<agrid, 32, 0, stream>>>(Qb, Kb, Vt, ctx);

    dim3 ogrid(DMODEL / 128, L_SEQ / 128);
    gemm_bf16_kernel<false, true><<<ogrid, 256, 0, stream>>>(
        ctx, woutb, d_out, bout, L_SEQ, DMODEL, DMODEL);
}
